// GAT_20469814133290
// MI455X (gfx1250) — matrix operand order repaired and checked by reading, not compiled, not run on hardware
//
#include <hip/hip_runtime.h>
#include <hip/hip_bf16.h>

typedef __attribute__((ext_vector_type(16))) _Float16 v16h;
typedef __attribute__((ext_vector_type(8)))  _Float16 v8h;
typedef __attribute__((ext_vector_type(8)))  float    v8f;

#define NEG_SLOPE 0.2f
#define EPS_F 1e-16f

static __device__ __forceinline__ void atomicMaxF(float* addr, float v) {
    // sign-split trick: works for IEEE ordering incl. -inf init
    if (v >= 0.0f) atomicMax((int*)addr, __float_as_int(v));
    else           atomicMin((unsigned int*)addr, __float_as_uint(v));
}

static __device__ __forceinline__ void edge_sd(const int* __restrict__ ei, int E,
                                               int idx, int& s, int& d) {
    if (idx < E) { s = ei[idx]; d = ei[E + idx]; }
    else         { s = idx - E; d = idx - E; }   // self loops appended
}

// ---------------- conversion / init kernels ----------------

__global__ void f32_to_f16_k(const float* __restrict__ src, _Float16* __restrict__ dst, long n) {
    long i = (long)blockIdx.x * blockDim.x + threadIdx.x;
    long st = (long)gridDim.x * blockDim.x;
    for (; i < n; i += st) dst[i] = (_Float16)src[i];
}

__global__ void relu_f32_to_f16_k(const float* __restrict__ src, _Float16* __restrict__ dst, long n) {
    long i = (long)blockIdx.x * blockDim.x + threadIdx.x;
    long st = (long)gridDim.x * blockDim.x;
    for (; i < n; i += st) { float v = src[i]; dst[i] = (_Float16)(v > 0.0f ? v : 0.0f); }
}

// W: K x Nc row-major (K=64).  Wt: NcPad x K row-major, zero padded cols.
__global__ void transpose_w_k(const float* __restrict__ W, _Float16* __restrict__ Wt,
                              int K, int Nc, int NcPad) {
    int i = blockIdx.x * blockDim.x + threadIdx.x;
    int tot = NcPad * K;
    if (i >= tot) return;
    int n = i / K, k = i - n * K;
    float v = (n < Nc) ? W[k * Nc + n] : 0.0f;
    Wt[(size_t)n * K + k] = (_Float16)v;
}

__global__ void init_ms_k(float* __restrict__ m, float* __restrict__ s, int n) {
    int i = blockIdx.x * blockDim.x + threadIdx.x;
    if (i < n) { m[i] = -__builtin_inff(); s[i] = 0.0f; }
}

__global__ void bias_init_k(float* __restrict__ out, const float* __restrict__ b, long total, int F) {
    long i = (long)blockIdx.x * blockDim.x + threadIdx.x;
    long st = (long)gridDim.x * blockDim.x;
    for (; i < total; i += st) out[i] = b[i % F];
}

// ---------------- WMMA GEMM:  C[M x NCOLREAL] = A[M x 64] * Bt^T  ----------------
// Bt is NCOLPAD x 64 (pre-transposed, zero-padded).  One wave owns a 16-row strip
// and iterates all column tiles, loading its A fragments exactly once.

template <int NCOLPAD, int NCOLREAL>
__global__ void __launch_bounds__(128)
wmma_gemm_f16_k(const _Float16* __restrict__ A, const _Float16* __restrict__ Bt,
                float* __restrict__ C, int M) {
    constexpr int NT = NCOLPAD / 16;           // column tiles per row strip
    const int lane = threadIdx.x & 31;
    const int wave = threadIdx.x >> 5;
    const int nRowTiles = M >> 4;
    const int rowTile = blockIdx.x * (blockDim.x >> 5) + wave;
    if (rowTile >= nRowTiles) return;          // wave-uniform: EXEC stays all-ones
    const int m0 = rowTile << 4;
    const int ln = lane & 15;
    const int hi = lane >> 4;

    // A fragments for both 32-wide K blocks (K=64), loaded once per row strip.
    // lane holds row m0+ln, K = hi*16 .. +15 contiguous per block.
    const _Float16* arow = A + (size_t)(m0 + ln) * 64 + hi * 16;
    v8h a00 = *(const v8h*)(arow);
    v8h a01 = *(const v8h*)(arow + 8);
    v8h a10 = *(const v8h*)(arow + 32);
    v8h a11 = *(const v8h*)(arow + 40);
    v16h a0, a1;
#pragma unroll
    for (int i = 0; i < 8; ++i) {
        a0[i] = a00[i]; a0[i + 8] = a01[i];
        a1[i] = a10[i]; a1[i + 8] = a11[i];
    }

#pragma unroll
    for (int nt = 0; nt < NT; ++nt) {
        // B: lane holds col nt*16+ln (Bt row), K = kb*32 + hi*16 + 0..15 contiguous
        const _Float16* brow = Bt + (size_t)(nt * 16 + ln) * 64 + hi * 16;
        v16h b0 = *(const v16h*)(brow);
        v16h b1 = *(const v16h*)(brow + 32);
        v8f c = {};
        c = __builtin_amdgcn_wmma_f32_16x16x32_f16(false, a0, false, b0, (short)0, c, false, false);
        c = __builtin_amdgcn_wmma_f32_16x16x32_f16(false, a1, false, b1, (short)0, c, false, false);

        const int col = nt * 16 + ln;
        if (col < NCOLREAL) {                  // guard padded cols (layer2: 48 -> 40)
            float* crow = C + (size_t)(m0 + hi * 8) * NCOLREAL + col;
#pragma unroll
            for (int j = 0; j < 8; ++j) crow[(size_t)j * NCOLREAL] = c[j];
        }
    }
}

// ---------------- attention ----------------

__global__ void attn_coeff_k(const float* __restrict__ H, const float* __restrict__ aw,
                             const float* __restrict__ dw, float* __restrict__ as_,
                             float* __restrict__ ad_, int M, int F) {
    int i = blockIdx.x * blockDim.x + threadIdx.x;
    if (i >= M) return;
    const float* h = H + (size_t)i * F;
    float s1 = 0.0f, s2 = 0.0f;
#pragma unroll 8
    for (int k = 0; k < F; ++k) { float v = h[k]; s1 += v * aw[k]; s2 += v * dw[k]; }
    as_[i] = s1; ad_[i] = s2;
}

__global__ void edge_max_k(const int* __restrict__ ei, int E, int Etot,
                           const float* __restrict__ as_, const float* __restrict__ ad_,
                           float* __restrict__ m) {
    int st = gridDim.x * blockDim.x;
    for (int idx = blockIdx.x * blockDim.x + threadIdx.x; idx < Etot; idx += st) {
        int s, d; edge_sd(ei, E, idx, s, d);
        float e = as_[s] + ad_[d];
        e = e > 0.0f ? e : NEG_SLOPE * e;
        atomicMaxF(m + d, e);
    }
}

__global__ void edge_sum_k(const int* __restrict__ ei, int E, int Etot,
                           const float* __restrict__ as_, const float* __restrict__ ad_,
                           const float* __restrict__ m, float* __restrict__ ssum) {
    int st = gridDim.x * blockDim.x;
    for (int idx = blockIdx.x * blockDim.x + threadIdx.x; idx < Etot; idx += st) {
        int s, d; edge_sd(ei, E, idx, s, d);
        float e = as_[s] + ad_[d];
        e = e > 0.0f ? e : NEG_SLOPE * e;
        atomicAdd(&ssum[d], __expf(e - m[d]));
    }
}

// one wave per edge; lane covers features lane and lane+32
__global__ void edge_agg_k(const int* __restrict__ ei, int E, int Etot,
                           const float* __restrict__ as_, const float* __restrict__ ad_,
                           const float* __restrict__ m, const float* __restrict__ ssum,
                           const float* __restrict__ H, float* __restrict__ out, int F) {
    int lane = threadIdx.x & 31;
    int wv   = (blockIdx.x * blockDim.x + threadIdx.x) >> 5;
    int nw   = (gridDim.x * blockDim.x) >> 5;
    for (int idx = wv; idx < Etot; idx += nw) {
        int s, d; edge_sd(ei, E, idx, s, d);
        float e = as_[s] + ad_[d];
        e = e > 0.0f ? e : NEG_SLOPE * e;
        float alpha = __expf(e - m[d]) / (ssum[d] + EPS_F);
        const float* hs = H + (size_t)s * F;
        float* od = out + (size_t)d * F;
        if (lane < F)      atomicAdd(od + lane,      alpha * hs[lane]);
        if (lane + 32 < F) atomicAdd(od + lane + 32, alpha * hs[lane + 32]);
    }
}

// ---------------- log_softmax over F=40, one wave per row, in place -------------

__global__ void log_softmax40_k(float* __restrict__ out, int M) {
    int lane = threadIdx.x & 31;
    int wv   = (blockIdx.x * blockDim.x + threadIdx.x) >> 5;
    int nw   = (gridDim.x * blockDim.x) >> 5;
    for (int row = wv; row < M; row += nw) {
        float* r = out + (size_t)row * 40;
        float v0 = r[lane];
        float v1 = (lane < 8) ? r[lane + 32] : -__builtin_inff();
        float mx = fmaxf(v0, v1);
        for (int o = 16; o > 0; o >>= 1) mx = fmaxf(mx, __shfl_xor(mx, o, 32));
        float se = __expf(v0 - mx) + ((lane < 8) ? __expf(v1 - mx) : 0.0f);
        for (int o = 16; o > 0; o >>= 1) se += __shfl_xor(se, o, 32);
        float l = __logf(se) + mx;
        r[lane] = v0 - l;
        if (lane < 8) r[lane + 32] = v1 - l;
    }
}

// ---------------- host launcher ----------------

static inline int cdiv_i(long a, int b) { return (int)((a + b - 1) / b); }

extern "C" void kernel_launch(void* const* d_in, const int* in_sizes, int n_in,
                              void* d_out, int out_size, void* d_ws, size_t ws_size,
                              hipStream_t stream) {
    const float* x    = (const float*)d_in[0];
    const int*   ei   = (const int*)  d_in[1];
    const float* W1   = (const float*)d_in[2];
    const float* aS1  = (const float*)d_in[3];
    const float* aD1  = (const float*)d_in[4];
    const float* b1   = (const float*)d_in[5];
    const float* W2   = (const float*)d_in[6];
    const float* aS2  = (const float*)d_in[7];
    const float* aD2  = (const float*)d_in[8];
    const float* b2   = (const float*)d_in[9];
    float* out = (float*)d_out;

    const int N    = in_sizes[0] / 64;     // 100000
    const int E    = in_sizes[1] / 2;      // 1600000
    const int Etot = E + N;

    // workspace carve-out (256B aligned); ~66 MB total with reuse
    char* w = (char*)d_ws;
    auto carve = [&](size_t bytes) {
        char* p = w; w += (bytes + 255) & ~(size_t)255; return (void*)p;
    };
    _Float16* xh  = (_Float16*)carve((size_t)N * 64 * 2);  // also reused for relu(h1) f16
    _Float16* w1t = (_Float16*)carve((size_t)64 * 64 * 2);
    _Float16* w2t = (_Float16*)carve((size_t)48 * 64 * 2);
    float*    h1  = (float*)carve((size_t)N * 64 * 4);     // also reused as h2 (N x 40)
    float*    o1  = (float*)carve((size_t)N * 64 * 4);
    float*    asv = (float*)carve((size_t)N * 4);
    float*    adv = (float*)carve((size_t)N * 4);
    float*    mv  = (float*)carve((size_t)N * 4);
    float*    sv  = (float*)carve((size_t)N * 4);

    const int T = 256;
    const long NF1 = (long)N * 64;
    const long NF2 = (long)N * 40;
    const int eBlocks = 2048;
    const int gemmBlocks = cdiv_i(N / 16, 4);  // 4 waves/block, one 16-row strip each

    // ---- prep ----
    f32_to_f16_k<<<4096, T, 0, stream>>>(x, xh, NF1);
    transpose_w_k<<<cdiv_i(64 * 64, T), T, 0, stream>>>(W1, w1t, 64, 64, 64);
    transpose_w_k<<<cdiv_i(48 * 64, T), T, 0, stream>>>(W2, w2t, 64, 40, 48);

    // ---- layer 1 ----
    wmma_gemm_f16_k<64, 64><<<gemmBlocks, 128, 0, stream>>>(xh, w1t, h1, N);
    attn_coeff_k<<<cdiv_i(N, T), T, 0, stream>>>(h1, aS1, aD1, asv, adv, N, 64);
    init_ms_k<<<cdiv_i(N, T), T, 0, stream>>>(mv, sv, N);
    bias_init_k<<<4096, T, 0, stream>>>(o1, b1, NF1, 64);
    edge_max_k<<<eBlocks, T, 0, stream>>>(ei, E, Etot, asv, adv, mv);
    edge_sum_k<<<eBlocks, T, 0, stream>>>(ei, E, Etot, asv, adv, mv, sv);
    edge_agg_k<<<eBlocks, T, 0, stream>>>(ei, E, Etot, asv, adv, mv, sv, h1, o1, 64);

    // ---- layer 2 (h2 reuses h1 buffer, relu(h1)_f16 reuses xh buffer) ----
    relu_f32_to_f16_k<<<4096, T, 0, stream>>>(o1, xh, NF1);
    wmma_gemm_f16_k<48, 40><<<gemmBlocks, 128, 0, stream>>>(xh, w2t, h1, N);
    attn_coeff_k<<<cdiv_i(N, T), T, 0, stream>>>(h1, aS2, aD2, asv, adv, N, 40);
    init_ms_k<<<cdiv_i(N, T), T, 0, stream>>>(mv, sv, N);
    bias_init_k<<<4096, T, 0, stream>>>(out, b2, NF2, 40);
    edge_max_k<<<eBlocks, T, 0, stream>>>(ei, E, Etot, asv, adv, mv);
    edge_sum_k<<<eBlocks, T, 0, stream>>>(ei, E, Etot, asv, adv, mv, sv);
    edge_agg_k<<<eBlocks, T, 0, stream>>>(ei, E, Etot, asv, adv, mv, sv, h1, out, 40);

    // ---- log_softmax ----
    log_softmax40_k<<<1024, T, 0, stream>>>(out, N);
}
